// Attention_13048110645532
// MI455X (gfx1250) — matrix operand order repaired and checked by reading, not compiled, not run on hardware
//
#include <hip/hip_runtime.h>
#include <hip/hip_bf16.h>

#define Bb 64
#define Ss 2048
#define Hh 1024
#define KK 2048   // 2*H, GEMM reduction dim

typedef __attribute__((ext_vector_type(16))) __bf16 v16bf;
typedef __attribute__((ext_vector_type(8)))  float  v8f;

union FragBF { uint4 u[2]; v16bf v; };

__device__ __forceinline__ unsigned short f2bf(float f) {
  unsigned u = __float_as_uint(f);
  u += 0x7FFFu + ((u >> 16) & 1u);   // round-to-nearest-even
  return (unsigned short)(u >> 16);
}

// pack two f32 -> packed bf16 dword; prefer single v_cvt_pk_bf16_f32
__device__ __forceinline__ unsigned pack2_bf16(float lo, float hi) {
#if __has_builtin(__builtin_amdgcn_cvt_pk_bf16_f32)
  typedef __attribute__((ext_vector_type(2))) __bf16 v2bf;
  union { v2bf v; unsigned u; } r;
  r.v = __builtin_amdgcn_cvt_pk_bf16_f32(lo, hi);
  return r.u;
#else
  return (unsigned)f2bf(lo) | ((unsigned)f2bf(hi) << 16);
#endif
}

__device__ __forceinline__ uint4 pack8_bf16(float4 a, float4 b) {
  uint4 r;
  r.x = pack2_bf16(a.x, a.y);
  r.y = pack2_bf16(a.z, a.w);
  r.z = pack2_bf16(b.x, b.y);
  r.w = pack2_bf16(b.z, b.w);
  return r;
}

__device__ __forceinline__ float fast_tanh(float x) {
  float ax = fabsf(x);
  float e  = __expf(-2.0f * ax);                       // v_exp_f32
  float t  = (1.0f - e) * __builtin_amdgcn_rcpf(1.0f + e);
  return copysignf(t, x);
}

// CDNA5 async global->LDS copy (ASYNCcnt-tracked). Low 32 bits of a generic
// LDS pointer == wave-relative LDS byte address (flat aperture mapping).
__device__ __forceinline__ void async_copy_b128(unsigned lds_off, const void* gptr) {
  asm volatile("global_load_async_to_lds_b128 %0, %1, off"
               :: "v"(lds_off), "v"(gptr)
               : "memory");
}
__device__ __forceinline__ void wait_asynccnt0() {
  asm volatile("s_wait_asynccnt 0x0" ::: "memory");
}

// ---------------------------------------------------------------------------
// Kernel 1: convert We = W_attn[H:3H, :] (row-major 2048x1024 f32) into bf16,
// tiled layout: we[(k/32)*1024 + n][k%32]  (each (kchunk,n) cell = 32 bf16)
// ---------------------------------------------------------------------------
__global__ void convert_we_kernel(const float* __restrict__ W_attn,
                                  unsigned short* __restrict__ we) {
  int g  = blockIdx.x * 256 + threadIdx.x;   // 0 .. 1024*1024-1
  int n  = g & (Hh - 1);
  int kp = g >> 10;                          // 0..1023
  int k0 = kp * 2;
  float f0 = W_attn[(size_t)(Hh + k0) * Hh + n];
  float f1 = W_attn[(size_t)(Hh + k0 + 1) * Hh + n];
  unsigned pack = pack2_bf16(f0, f1);
  size_t di = ((size_t)(k0 >> 5) * Hh + n) * 32 + (k0 & 31);
  *(unsigned*)&we[di] = pack;
}

// ---------------------------------------------------------------------------
// Kernel 2: hproj[b][n] = sum_h hidden[b][h] * Wh[h][n] + b_attn[n]
// ---------------------------------------------------------------------------
__global__ void hproj_kernel(const float* __restrict__ hidden,
                             const float* __restrict__ W_attn,
                             const float* __restrict__ b_attn,
                             float* __restrict__ hproj) {
  __shared__ float sh[Hh];
  int b = blockIdx.y;
  int n = blockIdx.x * 256 + threadIdx.x;
  for (int i = threadIdx.x; i < Hh; i += 256) sh[i] = hidden[(size_t)b * Hh + i];
  __syncthreads();
  float acc = b_attn[n];
#pragma unroll 8
  for (int h = 0; h < Hh; ++h) acc += sh[h] * W_attn[(size_t)h * Hh + n];
  hproj[(size_t)b * Hh + n] = acc;
}

// ---------------------------------------------------------------------------
// Kernel 3: fused GEMM(bf16 WMMA) + tanh + dot(v) -> partial scores (atomics)
// grid (H/128, S/64, B); 256 threads = 8 waves; BM=64, BN=128, BK=64,
// double-buffered LDS (explicit ping/pong), async global->LDS for B tile.
// ---------------------------------------------------------------------------
__global__ void __launch_bounds__(256)
attn_main_kernel(const float* __restrict__ enc,
                 const unsigned short* __restrict__ we,
                 const float* __restrict__ hproj,
                 const float* __restrict__ vvec,
                 float* __restrict__ scores) {
  // padded row stride 72 bf16 = 144B (16B-aligned, spreads banks)
  __shared__ __align__(16) unsigned short Ash[2][64 * 72];   // [buf][row s][k0..63]
  __shared__ __align__(16) unsigned short Bsh[2][128 * 72];  // [buf][col n][k0..63]

  const int b  = blockIdx.z;
  const int s0 = blockIdx.y * 64;
  const int n0 = blockIdx.x * 128;

  const int tid    = threadIdx.x;
  const int wave   = tid >> 5;
  const int lane   = tid & 31;
  const int wave_m = wave >> 1;    // 0..3 -> M sub-tile
  const int wave_n = wave & 1;     // 0..1 -> N half (64 cols)

  // A-fill mapping: row fr = tid&63, k-segment seg = tid>>6 (16 floats each)
  const int fr  = tid & 63;
  const int seg = tid >> 6;
  const float* arow = enc + ((size_t)b * Ss + (s0 + fr)) * (size_t)KK + seg * 16;

  // B-fill mapping: thread copies 64B: n_local = tid>>1, k-half bh = tid&1
  const int bn = tid >> 1;
  const int bh = tid & 1;

  v8f acc[4];
#pragma unroll
  for (int j = 0; j < 4; ++j)
#pragma unroll
    for (int i = 0; i < 8; ++i) acc[j][i] = 0.0f;

  // fragment read bases (ISA 16-bit A/B operand lane layouts)
  const int m   = lane & 15;
  const int khi = (lane >> 4) * 8;      // A: lane-half K offset
  const int kb2 = (lane >> 4) * 16;     // unused; B reads use khi offset too

  // --- stage B (two 32-K chunks starting at chunk c0) via async copies ---
  auto stageB = [&](int dbuf, int c0) {
    const unsigned short* sp0 = we + ((size_t)c0 * Hh + n0 + bn) * 32 + bh * 16;
    const unsigned short* sp1 = we + ((size_t)(c0 + 1) * Hh + n0 + bn) * 32 + bh * 16;
    unsigned d = (unsigned)(size_t)&Bsh[dbuf][bn * 72 + bh * 16];
    async_copy_b128(d,      sp0);
    async_copy_b128(d + 16, sp0 + 8);
    async_copy_b128(d + 64, sp1);       // k 32..63 cell (+32 elems)
    async_copy_b128(d + 80, sp1 + 8);
  };

  auto loadA = [&](int kk, float4* fa) {
    const float4* ep = (const float4*)(arow + kk);
    fa[0] = ep[0]; fa[1] = ep[1]; fa[2] = ep[2]; fa[3] = ep[3];
  };

  auto storeA = [&](int dbuf, const float4* fa) {
    uint4 w0 = pack8_bf16(fa[0], fa[1]);
    uint4 w1 = pack8_bf16(fa[2], fa[3]);
    *(uint4*)&Ash[dbuf][fr * 72 + seg * 16]     = w0;
    *(uint4*)&Ash[dbuf][fr * 72 + seg * 16 + 8] = w1;
  };

  auto computeTile = [&](int cbuf) {
#pragma unroll
    for (int kc = 0; kc < 2; ++kc) {
      FragBF a;
      const unsigned short* ar =
          &Ash[cbuf][(wave_m * 16 + m) * 72 + kc * 32 + khi];
      a.u[0] = *(const uint4*)ar;         // K = khi .. khi+7
      a.u[1] = *(const uint4*)(ar + 16);  // K = 16+khi .. 23+khi
#pragma unroll
      for (int j = 0; j < 4; ++j) {
        const int col = wave_n * 64 + j * 16 + m;
        const uint4* bp = (const uint4*)&Bsh[cbuf][col * 72 + kc * 32 + khi];
        FragBF bfr;
        bfr.u[0] = bp[0];                 // K = khi .. khi+7
        bfr.u[1] = bp[2];                 // K = 16+khi .. 23+khi
        acc[j] = __builtin_amdgcn_wmma_f32_16x16x32_bf16(
            false, a.v, false, bfr.v, (short)0, acc[j], false, false);
      }
    }
  };

  // ---- prologue: stage tile 0 into buffer 0 ----
  {
    float4 fa[4];
    loadA(0, fa);
    stageB(0, 0);
    storeA(0, fa);
    wait_asynccnt0();
  }
  __syncthreads();

  // ---- main pipeline: explicit ping/pong, 16 iterations, 16 WMMA each ----
  for (int kk = 0; kk < KK; kk += 128) {
    // ---- half 1: compute buf0 (tile kk), stage buf1 (tile kk+64) ----
    {
      float4 fa[4];
      loadA(kk + 64, fa);                 // issue global loads early
      stageB(1, (kk >> 5) + 2);           // async global->LDS
      __builtin_prefetch(arow + kk + 128, 0, 3);
      computeTile(0);                     // overlaps with outstanding loads
      storeA(1, fa);
      wait_asynccnt0();
    }
    __syncthreads();

    // ---- half 2: compute buf1 (tile kk+64), stage buf0 (tile kk+128) ----
    {
      const bool has_next = (kk + 128 < KK);
      float4 fa[4];
      if (has_next) {
        loadA(kk + 128, fa);
        stageB(0, (kk >> 5) + 4);
        __builtin_prefetch(arow + kk + 192, 0, 3);
      }
      computeTile(1);
      if (has_next) {
        storeA(0, fa);
        wait_asynccnt0();
      }
    }
    __syncthreads();
  }

  // ---- epilogue: tanh(acc + hproj) * v, reduce over N, atomic into scores
  const int nsub = lane & 15;
  const int half = lane >> 4;
  float rsum[8];
#pragma unroll
  for (int i = 0; i < 8; ++i) rsum[i] = 0.0f;

#pragma unroll
  for (int j = 0; j < 4; ++j) {
    int colg = n0 + wave_n * 64 + j * 16 + nsub;
    float hp = hproj[(size_t)b * Hh + colg];
    float vv = vvec[colg];
#pragma unroll
    for (int i = 0; i < 8; ++i)
      rsum[i] += fast_tanh(acc[j][i] + hp) * vv;
  }
#pragma unroll
  for (int i = 0; i < 8; ++i) {
    for (int off = 1; off < 16; off <<= 1)
      rsum[i] += __shfl_xor(rsum[i], off, 32);   // stays within 16-lane half
  }
  if (nsub == 0) {
#pragma unroll
    for (int i = 0; i < 8; ++i) {
      int row = s0 + wave_m * 16 + i + half * 8;  // C layout: M = i + 8*half
      atomicAdd(&scores[(size_t)b * Ss + row], rsum[i]);
    }
  }
}

// ---------------------------------------------------------------------------
// Kernel 4: mask + softmax over S per batch
// ---------------------------------------------------------------------------
__global__ void softmax_kernel(const float* __restrict__ scores,
                               const int* __restrict__ mask,
                               float* __restrict__ out) {
  __shared__ float red[256];
  int b = blockIdx.x;
  int tid = threadIdx.x;
  float xs[8];
  float lmax = -1e30f;
#pragma unroll
  for (int i = 0; i < 8; ++i) {
    int s = tid + i * 256;
    float x = scores[(size_t)b * Ss + s];
    if (mask[(size_t)b * Ss + s] == 0) x = -1.0e9f;
    xs[i] = x;
    lmax = fmaxf(lmax, x);
  }
  red[tid] = lmax;
  __syncthreads();
  for (int off = 128; off > 0; off >>= 1) {
    if (tid < off) red[tid] = fmaxf(red[tid], red[tid + off]);
    __syncthreads();
  }
  float bmax = red[0];
  __syncthreads();
  float lsum = 0.0f;
#pragma unroll
  for (int i = 0; i < 8; ++i) {
    float e = __expf(xs[i] - bmax);
    xs[i] = e;
    lsum += e;
  }
  red[tid] = lsum;
  __syncthreads();
  for (int off = 128; off > 0; off >>= 1) {
    if (tid < off) red[tid] += red[tid + off];
    __syncthreads();
  }
  float inv = __builtin_amdgcn_rcpf(red[0]);
#pragma unroll
  for (int i = 0; i < 8; ++i)
    out[(size_t)b * Ss + tid + i * 256] = xs[i] * inv;
}

// ---------------------------------------------------------------------------
extern "C" void kernel_launch(void* const* d_in, const int* in_sizes, int n_in,
                              void* d_out, int out_size, void* d_ws, size_t ws_size,
                              hipStream_t stream) {
  const float* hidden = (const float*)d_in[0];
  const float* enc    = (const float*)d_in[1];
  const int*   mask   = (const int*)d_in[2];
  const float* W_attn = (const float*)d_in[3];
  const float* b_attn = (const float*)d_in[4];
  const float* v      = (const float*)d_in[5];
  float* out = (float*)d_out;

  // workspace layout
  unsigned short* we = (unsigned short*)d_ws;                              // 4 MB
  float* hproj  = (float*)((char*)d_ws + (size_t)KK * Hh * 2);             // 256 KB
  float* scores = (float*)((char*)d_ws + (size_t)KK * Hh * 2
                                       + (size_t)Bb * Hh * 4);             // 512 KB

  hipMemsetAsync(scores, 0, (size_t)Bb * Ss * sizeof(float), stream);

  convert_we_kernel<<<dim3((KK / 2) * Hh / 256), 256, 0, stream>>>(W_attn, we);
  hproj_kernel<<<dim3(Hh / 256, Bb), 256, 0, stream>>>(hidden, W_attn, b_attn, hproj);
  attn_main_kernel<<<dim3(Hh / 128, Ss / 64, Bb), 256, 0, stream>>>(
      enc, we, hproj, v, scores);
  softmax_kernel<<<dim3(Bb), 256, 0, stream>>>(scores, mask, out);
}
